// TConformerQKV_61735859912778
// MI455X (gfx1250) — matrix operand order repaired and checked by reading, not compiled, not run on hardware
//
#include <hip/hip_runtime.h>
#include <math.h>

// ---------------- problem constants ----------------
#define NDIM  64
#define SEQT  512
#define NF    65
#define DFFN  256
#define WLEN  64
#define CHUNK 128          // query rows per block
#define NKROW 192          // key/value rows kept per block (window + chunk)
#define XS    66           // f32 LDS row stride for x / norm
#define KS    66           // bf16 LDS row stride for kh (row-major [s][c])
#define VS    200          // bf16 LDS row stride for vhT (transposed [c][s])

typedef __bf16 bf16_t;
typedef bf16_t v16bf __attribute__((ext_vector_type(16)));
typedef float  v8f   __attribute__((ext_vector_type(8)));

union Frag { v16bf v; unsigned int u[8]; };

// native bf16 conversion (RNE) -> v_cvt_*bf16_f32, not a software sequence
static __device__ __forceinline__ unsigned short f2bf(float x) {
  union { bf16_t h; unsigned short u; } c;
  c.h = (bf16_t)x;
  return c.u;
}
static __device__ __forceinline__ unsigned int pk2(float a, float b) {
  union { bf16_t h[2]; unsigned int u; } c;
  c.h[0] = (bf16_t)a;
  c.h[1] = (bf16_t)b;
  return c.u;
}
// A-matrix 16x32 bf16 lane/VGPR K offsets (contiguous, matches B)
static __device__ __forceinline__ int a_koff(int j, int half) {
  return 16 * half + 2 * j;
}
// B-matrix 32x16 bf16 lane/VGPR K offsets
static __device__ __forceinline__ int b_koff(int j, int half) {
  return 16 * half + 2 * j;
}
static __device__ __forceinline__ v8f wmma_bf(const Frag& a, const Frag& b, v8f c) {
  return __builtin_amdgcn_wmma_f32_16x16x32_bf16(false, a.v, false, b.v,
                                                 (short)0, c, false, false);
}
static __device__ __forceinline__ v8f vbroad(float x) {
  v8f r = {x, x, x, x, x, x, x, x};
  return r;
}
// A-fragment from channel-major global tensor: elem(ch,t) = base[(ch*SEQT+t)*NF]
static __device__ __forceinline__ Frag loadA_global(const float* base, int t, int kb, int half) {
  Frag f;
#pragma unroll
  for (int j = 0; j < 8; ++j) {
    int kk = kb + a_koff(j, half);
    f.u[j] = pk2(base[((size_t)kk * SEQT + t) * NF],
                 base[((size_t)(kk + 1) * SEQT + t) * NF]);
  }
  return f;
}
// B-fragment from row-major f32 weight W[n][k]: B[k][n] = W[nrow][kb+k]
static __device__ __forceinline__ Frag loadB_w(const float* W, int rs, int nrow, int kb, int half) {
  Frag f;
#pragma unroll
  for (int j = 0; j < 8; ++j) {
    const float* p = W + (size_t)nrow * rs + kb + b_koff(j, half);
    f.u[j] = pk2(p[0], p[1]);
  }
  return f;
}
// A-fragment from LDS f32 (converted to bf16 in-register)
static __device__ __forceinline__ Frag loadA_ldsf32(const float* s, int row, int kb, int half) {
  Frag f;
#pragma unroll
  for (int j = 0; j < 8; ++j) {
    const float* p = s + row * XS + kb + a_koff(j, half);
    f.u[j] = pk2(p[0], p[1]);
  }
  return f;
}
// A-fragment from LDS bf16 (stride even -> 4B aligned packed-pair reads)
static __device__ __forceinline__ Frag loadA_ldsbf(const unsigned short* s, int stride,
                                                   int row, int kb, int half) {
  Frag f;
#pragma unroll
  for (int j = 0; j < 8; ++j)
    f.u[j] = *(const unsigned int*)&s[row * stride + kb + a_koff(j, half)];
  return f;
}
static __device__ __forceinline__ float redmax16(float m) {
#pragma unroll
  for (int d = 1; d < 16; d <<= 1) m = fmaxf(m, __shfl_xor(m, d, 32));
  return m;
}
static __device__ __forceinline__ float redsum16(float s) {
#pragma unroll
  for (int d = 1; d < 16; d <<= 1) s += __shfl_xor(s, d, 32);
  return s;
}

__global__ __launch_bounds__(256) void conformer_fused_kernel(
    const float* __restrict__ qg, const float* __restrict__ kg, const float* __restrict__ vg,
    const float* __restrict__ inw, const float* __restrict__ inb,
    const float* __restrict__ outw, const float* __restrict__ outb,
    const float* __restrict__ l1w, const float* __restrict__ l1b,
    const float* __restrict__ w1,  const float* __restrict__ b1,
    const float* __restrict__ w2,  const float* __restrict__ b2,
    const float* __restrict__ l2w, const float* __restrict__ l2b,
    float* __restrict__ og)
{
  __shared__ __align__(16) float          xs[CHUNK * XS];      // x rows, later norm_out (f32)
  __shared__ __align__(16) unsigned short khs[NKROW * KS];     // K-heads bf16, [s][c]
  __shared__ __align__(16) unsigned short vhs[NDIM * VS];      // V-heads bf16, transposed [c][s]
  __shared__ __align__(16) unsigned short scr[8 * 3072];       // per-wave staging

  const int tid  = threadIdx.x;
  const int wave = tid >> 5;
  const int lane = tid & 31;
  const int half = lane >> 4;
  const int ln   = lane & 15;

  const int blk   = blockIdx.x;
  const int n     = blk >> 2;          // sequence index = b*NF + f
  const int chunk = blk & 3;
  const int bb    = n / NF;
  const int fi    = n % NF;
  const int q0    = chunk * CHUNK;
  const int kbase = q0 - WLEN;

  const size_t boff = (size_t)bb * NDIM * SEQT * NF + fi;
  const float* qx = qg + boff;
  const float* kx = kg + boff;
  const float* vx = vg + boff;

  // ================= Phase 1: K/V head projection + x copy =================
  for (int tile = wave; tile < NKROW / 16; tile += 8) {
    int s0 = kbase + tile * 16;
    if (s0 < 0) continue;                       // fully out-of-range key tile
    int t = s0 + ln;                            // A row = lane&15
    // ---- kh = kx @ Wk^T + bk  (Wk = rows [64,128) of in_proj_w)
    Frag a0 = loadA_global(kx, t, 0, half);
    Frag a1 = loadA_global(kx, t, 32, half);
#pragma unroll
    for (int ct = 0; ct < 4; ++ct) {
      v8f acc = vbroad(inb[NDIM + ct * 16 + ln]);
      Frag bA = loadB_w(inw, NDIM, NDIM + ct * 16 + ln, 0, half);
      Frag bB = loadB_w(inw, NDIM, NDIM + ct * 16 + ln, 32, half);
      acc = wmma_bf(a0, bA, acc);
      acc = wmma_bf(a1, bB, acc);
#pragma unroll
      for (int j = 0; j < 8; ++j)
        khs[(tile * 16 + j + 8 * half) * KS + ct * 16 + ln] = f2bf(acc[j]);
    }
    // ---- vh = vx @ Wv^T + bv  (Wv = rows [128,192)), stored TRANSPOSED [c][s]
    a0 = loadA_global(vx, t, 0, half);
    a1 = loadA_global(vx, t, 32, half);
#pragma unroll
    for (int ct = 0; ct < 4; ++ct) {
      v8f acc = vbroad(inb[2 * NDIM + ct * 16 + ln]);
      Frag bA = loadB_w(inw, NDIM, 2 * NDIM + ct * 16 + ln, 0, half);
      Frag bB = loadB_w(inw, NDIM, 2 * NDIM + ct * 16 + ln, 32, half);
      acc = wmma_bf(a0, bA, acc);
      acc = wmma_bf(a1, bB, acc);
#pragma unroll
      for (int j = 0; j < 8; ++j)
        vhs[(ct * 16 + ln) * VS + tile * 16 + j + 8 * half] = f2bf(acc[j]);
    }
  }
  // copy x rows for this block's query chunk (one 16-row tile per wave)
  {
    int t0 = q0 + wave * 16;
    for (int i = lane; i < 16 * NDIM; i += 32) {
      int r = i >> 6, ch = i & 63;
      xs[(wave * 16 + r) * XS + ch] = qx[((size_t)ch * SEQT + (t0 + r)) * NF];
    }
  }
  __syncthreads();   // only block-wide barrier; everything below is row-local

  // ================= Phase 2: one 16-row query tile per wave =================
  const int t0 = q0 + wave * 16;
  unsigned short* qst = scr + wave * 3072;   // 16x32 bf16 q stage
  unsigned short* Pst = qst + 512;           // 16x96 bf16 P stage (later 16x32 h1 stage)
  unsigned short* cst = Pst + 1536;          // 16x64 bf16 ctx / norm stage

  // zero K-pad columns once (qst cols 16..31, Pst cols 80..95)
  for (int i = lane; i < 256; i += 32) {
    qst[(i >> 4) * 32 + 16 + (i & 15)] = 0;
    Pst[(i >> 4) * 96 + 80 + (i & 15)] = 0;
  }

  // x A-fragments (K = channels, 2 chunks of 32)
  Frag xa0 = loadA_ldsf32(xs, wave * 16 + ln, 0, half);
  Frag xa1 = loadA_ldsf32(xs, wave * 16 + ln, 32, half);

  const int loclamp = (q0 == 0) ? WLEN : 0;   // lowest initialized vhT row pair

  // ---------------- attention: per head ----------------
#pragma unroll 1
  for (int h = 0; h < 4; ++h) {
    // qh tile = x @ Wq^T + bq  (fold 1/sqrt(Dh)=0.25 when staging)
    v8f qacc = vbroad(inb[h * 16 + ln]);
    {
      Frag bA = loadB_w(inw, NDIM, h * 16 + ln, 0, half);
      Frag bB = loadB_w(inw, NDIM, h * 16 + ln, 32, half);
      qacc = wmma_bf(xa0, bA, qacc);
      qacc = wmma_bf(xa1, bB, qacc);
    }
#pragma unroll
    for (int j = 0; j < 8; ++j)
      qst[(j + 8 * half) * 32 + ln] = f2bf(0.25f * qacc[j]);
    Frag qa = loadA_ldsbf(qst, 32, ln, 0, half);

    // scores vs 5 key tiles (window 79 fits in [t0-64, t0+16))
    v8f  sacc[5];
    bool val[5];
#pragma unroll
    for (int kt = 0; kt < 5; ++kt) {
      int s0 = t0 - WLEN + kt * 16;
      val[kt] = (s0 >= 0);
      if (!val[kt]) continue;
      Frag kb;
#pragma unroll
      for (int j = 0; j < 8; ++j) {
        if (half == 0)   // K = 2j,2j+1 (< Dh=16); upper half lanes carry the zero pad
          kb.u[j] = *(const unsigned int*)&khs[(s0 - kbase + ln) * KS + h * 16 + 2 * j];
        else
          kb.u[j] = 0u;
      }
      sacc[kt] = wmma_bf(qa, kb, vbroad(0.0f));
#pragma unroll
      for (int j = 0; j < 8; ++j) {
        int t = t0 + j + 8 * half;
        int s = s0 + ln;
        if (!((s <= t) && (s >= t - (WLEN - 1)))) sacc[kt][j] = -__builtin_inff();
      }
    }
    // softmax along the banded row (values of one row live in one 16-lane group)
    float mj[8], lj[8];
#pragma unroll
    for (int j = 0; j < 8; ++j) {
      float m = -__builtin_inff();
#pragma unroll
      for (int kt = 0; kt < 5; ++kt)
        if (val[kt]) m = fmaxf(m, sacc[kt][j]);
      mj[j] = redmax16(m);
    }
#pragma unroll
    for (int j = 0; j < 8; ++j) {
      float l = 0.0f;
#pragma unroll
      for (int kt = 0; kt < 5; ++kt)
        if (val[kt]) {
          float e = __expf(sacc[kt][j] - mj[j]);
          sacc[kt][j] = e;
          l += e;
        }
      lj[j] = 1.0f / redsum16(l);   // diagonal always unmasked -> l > 0
    }
    // stage P (bf16), zeros for invalid tiles
#pragma unroll
    for (int kt = 0; kt < 5; ++kt)
#pragma unroll
      for (int j = 0; j < 8; ++j)
        Pst[(j + 8 * half) * 96 + kt * 16 + ln] =
            val[kt] ? f2bf(sacc[kt][j] * lj[j]) : (unsigned short)0;

    // ctx = P @ V  (K = 96 with zero tail; vhT pair rows via single b32 loads)
    v8f cacc = vbroad(0.0f);
#pragma unroll
    for (int fr = 0; fr < 3; ++fr) {
      Frag pa = loadA_ldsbf(Pst, 96, ln, fr * 32, half);
      Frag vb;
#pragma unroll
      for (int j = 0; j < 8; ++j) {
        int idx = t0 - WLEN + fr * 32 + b_koff(j, half) - kbase;  // even
        idx = min(max(idx, loclamp), NKROW - 2);                  // clamped rows hit P==0 cols
        vb.u[j] = *(const unsigned int*)&vhs[(h * 16 + ln) * VS + idx];
      }
      cacc = wmma_bf(pa, vb, cacc);
    }
#pragma unroll
    for (int j = 0; j < 8; ++j)
      cst[(j + 8 * half) * 64 + h * 16 + ln] = f2bf(cacc[j]);
  }

  // ---------------- out-projection + residual + LN1 ----------------
  float yv[4][8];
  {
    Frag ca0 = loadA_ldsbf(cst, 64, ln, 0, half);
    Frag ca1 = loadA_ldsbf(cst, 64, ln, 32, half);
#pragma unroll
    for (int ct = 0; ct < 4; ++ct) {
      v8f acc = vbroad(outb[ct * 16 + ln]);
      Frag bA = loadB_w(outw, NDIM, ct * 16 + ln, 0, half);
      Frag bB = loadB_w(outw, NDIM, ct * 16 + ln, 32, half);
      acc = wmma_bf(ca0, bA, acc);
      acc = wmma_bf(ca1, bB, acc);
#pragma unroll
      for (int j = 0; j < 8; ++j)
        yv[ct][j] = acc[j] + xs[(wave * 16 + j + 8 * half) * XS + ct * 16 + ln];
    }
  }
  float muj[8], rsj[8];
#pragma unroll
  for (int j = 0; j < 8; ++j) {
    float s1 = yv[0][j] + yv[1][j] + yv[2][j] + yv[3][j];
    float s2 = yv[0][j] * yv[0][j] + yv[1][j] * yv[1][j] +
               yv[2][j] * yv[2][j] + yv[3][j] * yv[3][j];
    s1 = redsum16(s1);
    s2 = redsum16(s2);
    float mu = s1 * (1.0f / NDIM);
    float var = s2 * (1.0f / NDIM) - mu * mu;
    muj[j] = mu;
    rsj[j] = rsqrtf(var + 1e-5f);
  }
  {
    float gw[4], gb[4];
#pragma unroll
    for (int ct = 0; ct < 4; ++ct) { gw[ct] = l1w[ct * 16 + ln]; gb[ct] = l1b[ct * 16 + ln]; }
#pragma unroll
    for (int ct = 0; ct < 4; ++ct)
#pragma unroll
      for (int j = 0; j < 8; ++j) {
        float v = (yv[ct][j] - muj[j]) * rsj[j] * gw[ct] + gb[ct];
        xs[(wave * 16 + j + 8 * half) * XS + ct * 16 + ln] = v;  // norm_out (f32 residual)
        cst[(j + 8 * half) * 64 + ct * 16 + ln] = f2bf(v);       // norm_out (bf16 matmul A)
      }
  }

  // ---------------- FFN: ffw = gelu(norm @ w1^T + b1) @ w2^T + b2 ----------------
  {
    Frag na0 = loadA_ldsbf(cst, 64, ln, 0, half);
    Frag na1 = loadA_ldsbf(cst, 64, ln, 32, half);
    v8f facc[4];
#pragma unroll
    for (int ct = 0; ct < 4; ++ct) facc[ct] = vbroad(b2[ct * 16 + ln]);
    unsigned short* hst = Pst;   // reuse as 16x32 bf16 hidden stage
#pragma unroll 1
    for (int m = 0; m < DFFN / 32; ++m) {
#pragma unroll
      for (int u2 = 0; u2 < 2; ++u2) {
        int n0 = (2 * m + u2) * 16;
        v8f hh = vbroad(b1[n0 + ln]);
        Frag bA = loadB_w(w1, NDIM, n0 + ln, 0, half);
        Frag bB = loadB_w(w1, NDIM, n0 + ln, 32, half);
        hh = wmma_bf(na0, bA, hh);
        hh = wmma_bf(na1, bB, hh);
#pragma unroll
        for (int j = 0; j < 8; ++j) {
          float g = hh[j];
          g = 0.5f * g * (1.0f + erff(g * 0.70710678118654752f));  // exact gelu
          hst[(j + 8 * half) * 32 + u2 * 16 + ln] = f2bf(g);
        }
      }
      Frag ha = loadA_ldsbf(hst, 32, ln, 0, half);
#pragma unroll
      for (int ct = 0; ct < 4; ++ct) {
        Frag bW = loadB_w(w2, DFFN, ct * 16 + ln, 32 * m, half);
        facc[ct] = wmma_bf(ha, bW, facc[ct]);
      }
    }
#pragma unroll
    for (int ct = 0; ct < 4; ++ct)
#pragma unroll
      for (int j = 0; j < 8; ++j)
        yv[ct][j] = facc[ct][j] + xs[(wave * 16 + j + 8 * half) * XS + ct * 16 + ln];
  }

  // ---------------- LN2 + store (b f) t c -> b c t f ----------------
#pragma unroll
  for (int j = 0; j < 8; ++j) {
    float s1 = yv[0][j] + yv[1][j] + yv[2][j] + yv[3][j];
    float s2 = yv[0][j] * yv[0][j] + yv[1][j] * yv[1][j] +
               yv[2][j] * yv[2][j] + yv[3][j] * yv[3][j];
    s1 = redsum16(s1);
    s2 = redsum16(s2);
    float mu = s1 * (1.0f / NDIM);
    float var = s2 * (1.0f / NDIM) - mu * mu;
    muj[j] = mu;
    rsj[j] = rsqrtf(var + 1e-5f);
  }
  {
    float gw[4], gb[4];
#pragma unroll
    for (int ct = 0; ct < 4; ++ct) { gw[ct] = l2w[ct * 16 + ln]; gb[ct] = l2b[ct * 16 + ln]; }
#pragma unroll
    for (int ct = 0; ct < 4; ++ct)
#pragma unroll
      for (int j = 0; j < 8; ++j) {
        float r = (yv[ct][j] - muj[j]) * rsj[j] * gw[ct] + gb[ct];
        int c = ct * 16 + ln;
        int t = t0 + j + 8 * half;
        og[(((size_t)bb * NDIM + c) * SEQT + t) * NF + fi] = r;
      }
  }
}

extern "C" void kernel_launch(void* const* d_in, const int* in_sizes, int n_in,
                              void* d_out, int out_size, void* d_ws, size_t ws_size,
                              hipStream_t stream) {
  (void)n_in; (void)out_size; (void)d_ws; (void)ws_size;
  const int B = in_sizes[0] / (NDIM * SEQT * NF);   // batch
  dim3 grid(B * NF * (SEQT / CHUNK));
  dim3 block(256);
  conformer_fused_kernel<<<grid, block, 0, stream>>>(
      (const float*)d_in[0],  (const float*)d_in[1],  (const float*)d_in[2],
      (const float*)d_in[3],  (const float*)d_in[4],  (const float*)d_in[5],
      (const float*)d_in[6],  (const float*)d_in[7],  (const float*)d_in[8],
      (const float*)d_in[9],  (const float*)d_in[10], (const float*)d_in[11],
      (const float*)d_in[12], (const float*)d_in[13], (const float*)d_in[14],
      (float*)d_out);
}
